// MoE_MLP_37641093382344
// MI455X (gfx1250) — matrix operand order repaired and checked by reading, not compiled, not run on hardware
//
#include <hip/hip_runtime.h>
#include <hip/hip_bf16.h>
#include <math.h>

// Problem constants (from reference): B=4, N=2048 -> T=8192 tokens
#define T_TOK 8192
#define CDIM  1024
#define HDIM  4096
#define NEXP  8
#define TOPK  2

typedef __attribute__((ext_vector_type(16))) __bf16 v16bf;
typedef __attribute__((ext_vector_type(8)))  __bf16 v8bf;
typedef __attribute__((ext_vector_type(8)))  float  v8f;

static __device__ __forceinline__ unsigned short f32_to_bf16_rne(float f) {
  unsigned int u = __float_as_uint(f);
  u += 0x7FFFu + ((u >> 16) & 1u);
  return (unsigned short)(u >> 16);
}

// ---------------------------------------------------------------------------
// Kernel 0: zero the 8 expert counters
// ---------------------------------------------------------------------------
__global__ void moe_zero_counts(int* __restrict__ counts) {
  if (threadIdx.x < NEXP) counts[threadIdx.x] = 0;
}

// ---------------------------------------------------------------------------
// Kernel 1: gating. One wave (32 lanes) per token.
//  - logits[t,e] = x[t,:] . gate_w[:,e] + gate_b[e]  (cross-lane reduced)
//  - top-2 + softmax; record (expert, slot-in-expert, weight) per token
//  - append token to its two expert lists (atomic counters)
//  - cast x row to bf16 (coalesced stores)
// ---------------------------------------------------------------------------
__global__ __launch_bounds__(256)
void moe_gate(const float* __restrict__ x, const float* __restrict__ gw,
              const float* __restrict__ gb,
              int* __restrict__ counts, int* __restrict__ etok,
              int* __restrict__ tk_e, int* __restrict__ tk_p,
              float* __restrict__ tk_w, unsigned short* __restrict__ x_bf) {
  const int wave = threadIdx.x >> 5;
  const int lane = threadIdx.x & 31;
  const int t = blockIdx.x * 8 + wave;
  if (t >= T_TOK) return;

  const float* xr = x + (size_t)t * CDIM;
  float acc[NEXP];
#pragma unroll
  for (int e = 0; e < NEXP; ++e) acc[e] = 0.0f;

  for (int c = lane; c < CDIM; c += 32) {
    float xv = xr[c];
    x_bf[(size_t)t * CDIM + c] = f32_to_bf16_rne(xv);
    const float* gwr = gw + (size_t)c * NEXP;
#pragma unroll
    for (int e = 0; e < NEXP; ++e) acc[e] += xv * gwr[e];
  }
  // wave32 tree reduce each logit
#pragma unroll
  for (int e = 0; e < NEXP; ++e) {
#pragma unroll
    for (int off = 16; off >= 1; off >>= 1)
      acc[e] += __shfl_xor(acc[e], off, 32);
  }
  if (lane == 0) {
#pragma unroll
    for (int e = 0; e < NEXP; ++e) acc[e] += gb[e];
    int i0 = 0; float v0 = acc[0];
#pragma unroll
    for (int e = 1; e < NEXP; ++e)
      if (acc[e] > v0) { v0 = acc[e]; i0 = e; }     // ties -> lowest index
    int i1 = -1; float v1 = -3.4e38f;
#pragma unroll
    for (int e = 0; e < NEXP; ++e)
      if (e != i0 && acc[e] > v1) { v1 = acc[e]; i1 = e; }
    // softmax over the two kept logits
    float e1 = __expf(v1 - v0);
    float inv = 1.0f / (1.0f + e1);
    int p0 = atomicAdd(&counts[i0], 1);
    etok[i0 * T_TOK + p0] = t;
    int p1 = atomicAdd(&counts[i1], 1);
    etok[i1 * T_TOK + p1] = t;
    tk_e[t * 2 + 0] = i0; tk_p[t * 2 + 0] = p0; tk_w[t * 2 + 0] = inv;
    tk_e[t * 2 + 1] = i1; tk_p[t * 2 + 1] = p1; tk_w[t * 2 + 1] = e1 * inv;
  }
}

// ---------------------------------------------------------------------------
// Kernel 2: exclusive scan of the 8 counts -> packed row offsets
// ---------------------------------------------------------------------------
__global__ void moe_scan(const int* __restrict__ counts, int* __restrict__ rowoff) {
  if (threadIdx.x == 0 && blockIdx.x == 0) {
    int s = 0;
    for (int e = 0; e < NEXP; ++e) { rowoff[e] = s; s += counts[e]; }
    rowoff[NEXP] = s; // == T_TOK * TOPK
  }
}

// ---------------------------------------------------------------------------
// Kernel 3: cast weights fp32->bf16, pre-swizzled into WMMA B-fragment layout.
// Fragment (e, kt, nt): 32 lanes x 16 bf16, contiguous 1KB.
//   lane l, elem j  ->  K = kt*32 + (l>=16 ? 8 : 0) + (j&7) + (j>=8 ? 16 : 0),  N = nt*16 + (l&15)
// Consecutive nt fragments are contiguous -> a wave's 4 B-frags per K-step
// form one contiguous 4KB streak.
// ---------------------------------------------------------------------------
__global__ __launch_bounds__(256)
void moe_wconv(const float* __restrict__ w1, const float* __restrict__ w2,
               unsigned short* __restrict__ w1f, unsigned short* __restrict__ w2f) {
  long long i = (long long)blockIdx.x * blockDim.x + threadIdx.x;
  if (i >= (long long)NEXP * CDIM * HDIM) return;
  const int j = (int)(i & 15);
  const int l = (int)((i >> 4) & 31);
  long long rest = i >> 9;
  const int kq = ((l >> 4) << 3) + (j & 7) + ((j >> 3) << 4);   // K within the 32-slice
  const int n16 = l & 15;
  { // w1: [E, C, H], K-dim = C, N-dim = H
    int nt = (int)(rest % (HDIM / 16));
    long long r2 = rest / (HDIM / 16);
    int kt = (int)(r2 % (CDIM / 32));
    int e  = (int)(r2 / (CDIM / 32));
    int k = kt * 32 + kq, n = nt * 16 + n16;
    w1f[i] = f32_to_bf16_rne(w1[((size_t)e * CDIM + k) * HDIM + n]);
  }
  { // w2: [E, H, C], K-dim = H, N-dim = C
    int nt = (int)(rest % (CDIM / 16));
    long long r2 = rest / (CDIM / 16);
    int kt = (int)(r2 % (HDIM / 32));
    int e  = (int)(r2 / (HDIM / 32));
    int k = kt * 32 + kq, n = nt * 16 + n16;
    w2f[i] = f32_to_bf16_rne(w2[((size_t)e * HDIM + k) * CDIM + n]);
  }
}

// A-fragment loader: row-major bf16 row, ISA 16-bit A 16x32 layout:
// lane l holds row (l&15); elems 0..7 = K0+(0..7), elems 8..15 = K0+16+(0..7)
// with K0 = (l>=16 ? 8 : 0).
static __device__ __forceinline__ v16bf load_afrag(const unsigned short* p) {
  v8bf lo = *(const v8bf*)(p);
  v8bf hi = *(const v8bf*)(p + 16);
  return __builtin_shufflevector(lo, hi, 0,1,2,3,4,5,6,7,8,9,10,11,12,13,14,15);
}

static __device__ __forceinline__ v8f wmma_bf16(v16bf a, v16bf b, v8f c) {
  return __builtin_amdgcn_wmma_f32_16x16x32_bf16(
      false, a, false, b, (short)0, c, false, false);
}

// ---------------------------------------------------------------------------
// Kernel 4: GEMM1  h = gelu(x[tok] @ w1[e] + b1[e])
// Register-blocked: each wave computes a 32x64 tile = 2 M-frags x 4 N-frags,
// 8 WMMAs per K-step on 12 b128 loads (vs 4 loads/WMMA unblocked).
// Block = 128 threads (4 waves along N) -> block tile 32 x 256.
// grid.x = E * T/32 row tiles (early exit past expert count), grid.y = H/256.
// ---------------------------------------------------------------------------
__global__ __launch_bounds__(128)
void moe_gemm1(const unsigned short* __restrict__ x_bf,
               const unsigned short* __restrict__ w1f,
               const float* __restrict__ b1,
               const int* __restrict__ counts, const int* __restrict__ rowoff,
               const int* __restrict__ etok, unsigned short* __restrict__ hbuf) {
  const int e   = blockIdx.x / (T_TOK / 32);
  const int rt  = blockIdx.x % (T_TOK / 32);
  const int cnt = counts[e];
  const int row0 = rt * 32;
  if (row0 >= cnt) return;

  const int wave = threadIdx.x >> 5;
  const int lane = threadIdx.x & 31;
  const int nt0  = (blockIdx.y * 4 + wave) * 4;     // first 16-col frag of 4

  const int m = lane & 15;
  const int ra = (row0 + m      < cnt) ? (row0 + m)      : (cnt - 1);
  const int rb = (row0 + 16 + m < cnt) ? (row0 + 16 + m) : (cnt - 1);
  const int tokA = etok[e * T_TOK + ra];
  const int tokB = etok[e * T_TOK + rb];
  const unsigned short* arowA = x_bf + (size_t)tokA * CDIM + ((lane >> 4) << 3);
  const unsigned short* arowB = x_bf + (size_t)tokB * CDIM + ((lane >> 4) << 3);
  const unsigned short* bfrag = w1f +
      ((size_t)e * (CDIM / 32) * (HDIM / 16) + nt0) * 512 + lane * 16;
  const size_t bstep = (size_t)(HDIM / 16) * 512;

  v8f acc[2][4] = {};
#pragma unroll 2
  for (int kt = 0; kt < CDIM / 32; ++kt) {
    __builtin_prefetch((const void*)(bfrag + (kt + 1) * bstep), 0, 1);
    v16bf aA = load_afrag(arowA + kt * 32);
    v16bf aB = load_afrag(arowB + kt * 32);
    v16bf b0 = *(const v16bf*)(bfrag + kt * bstep);
    v16bf b1v = *(const v16bf*)(bfrag + kt * bstep + 512);
    v16bf b2v = *(const v16bf*)(bfrag + kt * bstep + 1024);
    v16bf b3 = *(const v16bf*)(bfrag + kt * bstep + 1536);
    acc[0][0] = wmma_bf16(aA, b0, acc[0][0]);
    acc[0][1] = wmma_bf16(aA, b1v, acc[0][1]);
    acc[0][2] = wmma_bf16(aA, b2v, acc[0][2]);
    acc[0][3] = wmma_bf16(aA, b3, acc[0][3]);
    acc[1][0] = wmma_bf16(aB, b0, acc[1][0]);
    acc[1][1] = wmma_bf16(aB, b1v, acc[1][1]);
    acc[1][2] = wmma_bf16(aB, b2v, acc[1][2]);
    acc[1][3] = wmma_bf16(aB, b3, acc[1][3]);
  }

  const int hbase = rowoff[e] + row0;
#pragma unroll
  for (int f = 0; f < 4; ++f) {
    const int col = (nt0 + f) * 16 + (lane & 15);
    const float bias = b1[e * HDIM + col];
#pragma unroll
    for (int mf = 0; mf < 2; ++mf) {
#pragma unroll
      for (int r = 0; r < 8; ++r) {
        const int mm = mf * 16 + r + ((lane >> 4) << 3);  // C/D: M = r + 8*(l>=16)
        if (row0 + mm < cnt) {
          float v = acc[mf][f][r] + bias;
          float g = 0.5f * v * (1.0f + erff(v * 0.70710678f));  // exact GELU
          hbuf[(size_t)(hbase + mm) * HDIM + col] = f32_to_bf16_rne(g);
        }
      }
    }
  }
}

// ---------------------------------------------------------------------------
// Kernel 5: GEMM2  y[row] = h[row] @ w2[e] + b2[e]   (unscaled, packed rows)
// Same 32x64-per-wave register blocking. No atomics: the combine kernel
// applies gating weights afterwards.
// ---------------------------------------------------------------------------
__global__ __launch_bounds__(128)
void moe_gemm2(const unsigned short* __restrict__ hbuf,
               const unsigned short* __restrict__ w2f,
               const float* __restrict__ b2,
               const int* __restrict__ counts, const int* __restrict__ rowoff,
               float* __restrict__ ybuf) {
  const int e   = blockIdx.x / (T_TOK / 32);
  const int rt  = blockIdx.x % (T_TOK / 32);
  const int cnt = counts[e];
  const int row0 = rt * 32;
  if (row0 >= cnt) return;

  const int wave = threadIdx.x >> 5;
  const int lane = threadIdx.x & 31;
  const int nt0  = (blockIdx.y * 4 + wave) * 4;     // first 16-col frag of 4

  const int m = lane & 15;
  const int ra = (row0 + m      < cnt) ? (row0 + m)      : (cnt - 1);
  const int rb = (row0 + 16 + m < cnt) ? (row0 + 16 + m) : (cnt - 1);
  const unsigned short* arowA = hbuf +
      (size_t)(rowoff[e] + ra) * HDIM + ((lane >> 4) << 3);
  const unsigned short* arowB = hbuf +
      (size_t)(rowoff[e] + rb) * HDIM + ((lane >> 4) << 3);
  const unsigned short* bfrag = w2f +
      ((size_t)e * (HDIM / 32) * (CDIM / 16) + nt0) * 512 + lane * 16;
  const size_t bstep = (size_t)(CDIM / 16) * 512;

  v8f acc[2][4] = {};
#pragma unroll 2
  for (int kt = 0; kt < HDIM / 32; ++kt) {
    __builtin_prefetch((const void*)(bfrag + (kt + 1) * bstep), 0, 1);
    v16bf aA = load_afrag(arowA + kt * 32);
    v16bf aB = load_afrag(arowB + kt * 32);
    v16bf b0 = *(const v16bf*)(bfrag + kt * bstep);
    v16bf b1v = *(const v16bf*)(bfrag + kt * bstep + 512);
    v16bf b2v = *(const v16bf*)(bfrag + kt * bstep + 1024);
    v16bf b3 = *(const v16bf*)(bfrag + kt * bstep + 1536);
    acc[0][0] = wmma_bf16(aA, b0, acc[0][0]);
    acc[0][1] = wmma_bf16(aA, b1v, acc[0][1]);
    acc[0][2] = wmma_bf16(aA, b2v, acc[0][2]);
    acc[0][3] = wmma_bf16(aA, b3, acc[0][3]);
    acc[1][0] = wmma_bf16(aB, b0, acc[1][0]);
    acc[1][1] = wmma_bf16(aB, b1v, acc[1][1]);
    acc[1][2] = wmma_bf16(aB, b2v, acc[1][2]);
    acc[1][3] = wmma_bf16(aB, b3, acc[1][3]);
  }

  const int ybase = rowoff[e] + row0;
#pragma unroll
  for (int f = 0; f < 4; ++f) {
    const int col = (nt0 + f) * 16 + (lane & 15);
    const float bias = b2[e * CDIM + col];
#pragma unroll
    for (int mf = 0; mf < 2; ++mf) {
#pragma unroll
      for (int r = 0; r < 8; ++r) {
        const int mm = mf * 16 + r + ((lane >> 4) << 3);
        if (row0 + mm < cnt)
          ybuf[(size_t)(ybase + mm) * CDIM + col] = acc[mf][f][r] + bias;
      }
    }
  }
}

// ---------------------------------------------------------------------------
// Kernel 6: out[t,:] = w0 * y[r0,:] + w1 * y[r1,:]  (float4, pure bandwidth)
// ---------------------------------------------------------------------------
__global__ __launch_bounds__(256)
void moe_combine(const float* __restrict__ ybuf, const int* __restrict__ rowoff,
                 const int* __restrict__ tk_e, const int* __restrict__ tk_p,
                 const float* __restrict__ tk_w, float* __restrict__ out) {
  const int i = blockIdx.x * blockDim.x + threadIdx.x;
  if (i >= T_TOK * (CDIM / 4)) return;
  const int t  = i / (CDIM / 4);
  const int c4 = i % (CDIM / 4);
  const int r0 = rowoff[tk_e[t * 2 + 0]] + tk_p[t * 2 + 0];
  const int r1 = rowoff[tk_e[t * 2 + 1]] + tk_p[t * 2 + 1];
  const float w0 = tk_w[t * 2 + 0];
  const float w1 = tk_w[t * 2 + 1];
  const float4 ya = ((const float4*)ybuf)[(size_t)r0 * (CDIM / 4) + c4];
  const float4 yb = ((const float4*)ybuf)[(size_t)r1 * (CDIM / 4) + c4];
  float4 o;
  o.x = w0 * ya.x + w1 * yb.x;
  o.y = w0 * ya.y + w1 * yb.y;
  o.z = w0 * ya.z + w1 * yb.z;
  o.w = w0 * ya.w + w1 * yb.w;
  ((float4*)out)[(size_t)t * (CDIM / 4) + c4] = o;
}

// ---------------------------------------------------------------------------
extern "C" void kernel_launch(void* const* d_in, const int* in_sizes, int n_in,
                              void* d_out, int out_size, void* d_ws, size_t ws_size,
                              hipStream_t stream) {
  (void)in_sizes; (void)n_in; (void)out_size; (void)ws_size;
  const float* x      = (const float*)d_in[0];
  const float* gate_w = (const float*)d_in[1];
  const float* gate_b = (const float*)d_in[2];
  const float* w1     = (const float*)d_in[3];
  const float* b1     = (const float*)d_in[4];
  const float* w2     = (const float*)d_in[5];
  const float* b2     = (const float*)d_in[6];
  float* out = (float*)d_out;

  // Workspace layout (all chunks multiples of 256 B)
  char* p = (char*)d_ws;
  int*   etok           = (int*)p;            p += (size_t)NEXP * T_TOK * 4;   // 256 KB
  int*   tk_e           = (int*)p;            p += (size_t)T_TOK * 2 * 4;      // 64 KB
  int*   tk_p           = (int*)p;            p += (size_t)T_TOK * 2 * 4;      // 64 KB
  float* tk_w           = (float*)p;          p += (size_t)T_TOK * 2 * 4;      // 64 KB
  int*   counts         = (int*)p;            p += 256;
  int*   rowoff         = (int*)p;            p += 256;
  unsigned short* x_bf  = (unsigned short*)p; p += (size_t)T_TOK * CDIM * 2;   // 16 MB
  unsigned short* w1f   = (unsigned short*)p; p += (size_t)NEXP * CDIM * HDIM * 2; // 64 MB
  unsigned short* w2f   = (unsigned short*)p; p += (size_t)NEXP * HDIM * CDIM * 2; // 64 MB
  unsigned short* hbuf  = (unsigned short*)p; p += (size_t)T_TOK * TOPK * HDIM * 2; // 128 MB
  float* ybuf           = (float*)p;          // T*K rows x C fp32 = 64 MB

  moe_zero_counts<<<1, 32, 0, stream>>>(counts);
  moe_gate<<<T_TOK / 8, 256, 0, stream>>>(x, gate_w, gate_b, counts, etok,
                                          tk_e, tk_p, tk_w, x_bf);
  moe_scan<<<1, 32, 0, stream>>>(counts, rowoff);
  {
    long long total = (long long)NEXP * CDIM * HDIM;
    moe_wconv<<<(unsigned)((total + 255) / 256), 256, 0, stream>>>(w1, w2, w1f, w2f);
  }
  dim3 g1(NEXP * (T_TOK / 32), HDIM / 256);
  moe_gemm1<<<g1, 128, 0, stream>>>(x_bf, w1f, b1, counts, rowoff, etok, hbuf);
  dim3 g2(NEXP * (T_TOK / 32), CDIM / 256);
  moe_gemm2<<<g2, 128, 0, stream>>>(hbuf, w2f, b2, counts, rowoff, ybuf);
  moe_combine<<<(T_TOK * (CDIM / 4) + 255) / 256, 256, 0, stream>>>(
      ybuf, rowoff, tk_e, tk_p, tk_w, out);
}
